// RandLANet_81870666596688
// MI455X (gfx1250) — matrix operand order repaired and checked by reading, not compiled, not run on hardware
//
#include <hip/hip_runtime.h>
#include <hip/hip_bf16.h>
#include <math.h>

typedef _Float16 v16h __attribute__((ext_vector_type(16)));
typedef float    v8f  __attribute__((ext_vector_type(8)));

#define BN_S 0.9999995f
#define NPTS 8192
#define NB   2

#define ACT_NONE  0
#define ACT_RELU  1
#define ACT_LRELU 2

// LDS capacities (elements). Max over all layers:
//   W: CoutPad*(KPad+8): bottleneck 128*(128+8)=17408 ; dec0 64*(256+8)=16896
//   A: 16*(256+8)=4224
#define SW_ELEMS 17408
#define SA_ELEMS 4224

// ---------------------------------------------------------------------------
// Fused linear: out[r, off+o] = act(scale * sum_c in[r,c]*W[o,c] + bias[o])
// One block (8 wave32) = one 16-row M tile; wave w computes N tile w.
// W and A staged zero-padded in LDS (clamped unconditional loads -> no exec
// branching); inner loop is pure ds_load + v_wmma_f32_16x16x32_f16.
// ---------------------------------------------------------------------------
__global__ __launch_bounds__(256) void k_linear_wmma(
    const float* __restrict__ in, int inStride,
    const float* __restrict__ W,      // (Cout, Cin) row-major
    const float* __restrict__ bias,   // may be nullptr
    float* __restrict__ out, int outStride, int outOff,
    int R, int Cin, int Cout, int act, float scale)
{
    __shared__ _Float16 sW[SW_ELEMS];
    __shared__ _Float16 sA[SA_ELEMS];

    const int KPad    = (Cin + 31) & ~31;
    const int pitch   = KPad + 8;                 // bank-conflict-free row pitch
    const int CoutPad = (Cout + 15) & ~15;
    const int tilesN  = CoutPad >> 4;
    const int tm      = blockIdx.x;               // M tile

    // ---- stage W (zero-padded), clamped loads: no exec-mask branching ----
    const int wTot = CoutPad * pitch;
    for (int i = threadIdx.x; i < wTot; i += 256) {
        const int o = i / pitch, k = i % pitch;
        const int oc = o < Cout ? o : Cout - 1;
        const int kc = k < Cin ? k : Cin - 1;
        const float v = W[(size_t)oc * Cin + kc];
        sW[i] = (o < Cout && k < Cin) ? (_Float16)v : (_Float16)0.0f;
    }
    // ---- stage A tile (16 rows, zero-padded) ----
    const int aTot = 16 * pitch;
    for (int i = threadIdx.x; i < aTot; i += 256) {
        const int r = i / pitch, k = i % pitch;
        const int row = tm * 16 + r;
        const int rc = row < R ? row : R - 1;
        const int kc = k < Cin ? k : Cin - 1;
        const float v = in[(size_t)rc * inStride + kc];
        sA[i] = (row < R && k < Cin) ? (_Float16)v : (_Float16)0.0f;
    }
    __syncthreads();

    const int wib  = threadIdx.x >> 5;
    const int lane = threadIdx.x & 31;
    const int mloc = lane & 15;
    const int hiK  = lane >> 4;

    for (int tn = wib; tn < tilesN; tn += 8) {    // wave-uniform loop
        v8f acc = {};
        const _Float16* aRow = sA + mloc * pitch;
        const _Float16* bRow = sW + (tn * 16 + mloc) * pitch;
        for (int k0 = 0; k0 < KPad; k0 += 32) {
            v16h a, b;
#pragma unroll
            for (int v = 0; v < 8; ++v) {
#pragma unroll
                for (int h = 0; h < 2; ++h) {
                    // A 16x32 f16: K = (v/4)*16 + hiK*8 + (v%4)*2 + h (2 runs of 8 contiguous)
                    a[v * 2 + h] = aRow[k0 + ((v >> 2) << 4) + (hiK << 3) + ((v & 3) << 1) + h];
                    // B 32x16 f16: K = (v/4)*16 + hiK*8 + (v%4)*2 + h (2 runs of 8 contiguous)
                    b[v * 2 + h] = bRow[k0 + ((v >> 2) << 4) + (hiK << 3) + ((v & 3) << 1) + h];
                }
            }
            acc = __builtin_amdgcn_wmma_f32_16x16x32_f16(false, a, false, b,
                                                         (short)0, acc, false, false);
        }
        const int nCol = tn * 16 + mloc;
#pragma unroll
        for (int r = 0; r < 8; ++r) {
            const int mo = tm * 16 + r + (hiK << 3);
            if (mo < R && nCol < Cout) {
                float v = acc[r] * scale;
                if (bias) v += bias[nCol];
                if      (act == ACT_RELU)  v = v > 0.0f ? v : 0.0f;
                else if (act == ACT_LRELU) v = v > 0.0f ? v : 0.2f * v;
                out[(size_t)mo * outStride + outOff + nCol] = v;
            }
        }
    }
}

// ---------------------------------------------------------------------------
// Brute-force 16-NN with LDS point tiling. grid = (ceil(n/256), NB)
// ---------------------------------------------------------------------------
__global__ __launch_bounds__(256) void k_knn16(const float* __restrict__ pts,
                                               int n, int* __restrict__ idx)
{
    __shared__ float sx[256], sy[256], sz[256];
    const int b = blockIdx.y;
    const int i = blockIdx.x * 256 + threadIdx.x;
    const float* base = pts + (size_t)b * NPTS * 6;
    const bool active = i < n;
    const int ic = active ? i : 0;
    const float xi = base[ic * 6 + 0];
    const float yi = base[ic * 6 + 1];
    const float zi = base[ic * 6 + 2];

    float best[16]; int bid[16];
#pragma unroll
    for (int s = 0; s < 16; ++s) { best[s] = 3.4e38f; bid[s] = 0; }

    for (int j0 = 0; j0 < n; j0 += 256) {
        const int cnt = (n - j0) < 256 ? (n - j0) : 256;
        __syncthreads();
        if (threadIdx.x < cnt) {
            const int j = j0 + threadIdx.x;
            sx[threadIdx.x] = base[j * 6 + 0];
            sy[threadIdx.x] = base[j * 6 + 1];
            sz[threadIdx.x] = base[j * 6 + 2];
        }
        __syncthreads();
        for (int c = 0; c < cnt; ++c) {
            const float dx = xi - sx[c], dy = yi - sy[c], dz = zi - sz[c];
            const float d = dx * dx + dy * dy + dz * dz;
            if (d < best[15]) {
                float dd = d; int ii = j0 + c;
#pragma unroll
                for (int s = 0; s < 16; ++s) {
                    if (dd < best[s]) {
                        const float td = best[s]; const int ti = bid[s];
                        best[s] = dd; bid[s] = ii; dd = td; ii = ti;
                    }
                }
            }
        }
    }
    if (active) {
        int* o = idx + ((size_t)b * n + i) * 16;
#pragma unroll
        for (int s = 0; s < 16; ++s) o[s] = bid[s];
    }
}

// ---------------------------------------------------------------------------
// Relative position encoding: [ctr(3), nb(3), diff(3), |diff|] -> (B,n,16,10)
// ---------------------------------------------------------------------------
__global__ void k_build_rel(const float* __restrict__ pts, const int* __restrict__ idx,
                            int n, float* __restrict__ rel)
{
    const int t = blockIdx.x * blockDim.x + threadIdx.x;
    if (t >= NB * n * 16) return;
    const int b   = t / (n * 16);
    const int rem = t % (n * 16);
    const int i   = rem / 16;
    const float* base = pts + (size_t)b * NPTS * 6;
    const int j = idx[t];
    const float cx = base[i * 6 + 0], cy = base[i * 6 + 1], cz = base[i * 6 + 2];
    const float nx = base[j * 6 + 0], ny = base[j * 6 + 1], nz = base[j * 6 + 2];
    const float dx = cx - nx, dy = cy - ny, dz = cz - nz;
    float* r = rel + (size_t)t * 10;
    r[0] = cx; r[1] = cy; r[2] = cz;
    r[3] = nx; r[4] = ny; r[5] = nz;
    r[6] = dx; r[7] = dy; r[8] = dz;
    r[9] = sqrtf(dx * dx + dy * dy + dz * dz);
}

// xcat[row, c0+c] = feat[row/16, c]   rows = B*n*16
__global__ void k_concat_fe(float* __restrict__ xcat, int dtot, int c0,
                            const float* __restrict__ feat, int fc, int rows)
{
    const int t = blockIdx.x * blockDim.x + threadIdx.x;
    if (t >= rows * fc) return;
    const int row = t / fc, c = t % fc;
    xcat[(size_t)row * dtot + c0 + c] = feat[(size_t)(row / 16) * fc + c];
}

// softmax over 16 neighbors (per row, per channel) + weighted sum
__global__ void k_softmax_pool(const float* __restrict__ x, const float* __restrict__ s,
                               int Rp, int C, float* __restrict__ agg)
{
    const int t = blockIdx.x * blockDim.x + threadIdx.x;
    if (t >= Rp * C) return;
    const int r = t / C, c = t % C;
    const float* xs = x + (size_t)r * 16 * C + c;
    const float* ss = s + (size_t)r * 16 * C + c;
    float mx = -3.4e38f;
#pragma unroll
    for (int k = 0; k < 16; ++k) { const float v = ss[k * C]; mx = v > mx ? v : mx; }
    float sum = 0.0f, acc = 0.0f;
#pragma unroll
    for (int k = 0; k < 16; ++k) {
        const float e = expf(ss[k * C] - mx);
        sum += e; acc += e * xs[k * C];
    }
    agg[t] = acc / sum;
}

__global__ void k_add_lrelu(const float* __restrict__ a, const float* __restrict__ b,
                            float* __restrict__ dst, int cnt)
{
    const int t = blockIdx.x * blockDim.x + threadIdx.x;
    if (t >= cnt) return;
    const float v = a[t] + b[t];
    dst[t] = v > 0.0f ? v : 0.2f * v;
}

// row prefix per batch: (B, n_old, C) -> (B, n_new, C)
__global__ void k_subsample(const float* __restrict__ src, float* __restrict__ dst,
                            int n_old, int n_new, int C)
{
    const int t = blockIdx.x * blockDim.x + threadIdx.x;
    if (t >= NB * n_new * C) return;
    const int c = t % C;
    const int i = (t / C) % n_new;
    const int b = t / (C * n_new);
    dst[t] = src[((size_t)b * n_old + i) * C + c];
}

// ---------------------------------------------------------------------------
// 3-NN over prefix points + inverse-dist weights; LDS tiled. grid=(ceil(n_dn/256),NB)
// ---------------------------------------------------------------------------
__global__ __launch_bounds__(256) void k_find3(const float* __restrict__ pts,
                                               int n_dn, int n_sp,
                                               int* __restrict__ idx3,
                                               float* __restrict__ w3)
{
    __shared__ float sx[256], sy[256], sz[256];
    const int b = blockIdx.y;
    const int i = blockIdx.x * 256 + threadIdx.x;
    const float* base = pts + (size_t)b * NPTS * 6;
    const bool active = i < n_dn;
    const int ic = active ? i : 0;
    const float xi = base[ic * 6 + 0];
    const float yi = base[ic * 6 + 1];
    const float zi = base[ic * 6 + 2];

    float bd[3] = {3.4e38f, 3.4e38f, 3.4e38f};
    int   bi[3] = {0, 0, 0};

    for (int j0 = 0; j0 < n_sp; j0 += 256) {
        const int cnt = (n_sp - j0) < 256 ? (n_sp - j0) : 256;
        __syncthreads();
        if (threadIdx.x < cnt) {
            const int j = j0 + threadIdx.x;
            sx[threadIdx.x] = base[j * 6 + 0];
            sy[threadIdx.x] = base[j * 6 + 1];
            sz[threadIdx.x] = base[j * 6 + 2];
        }
        __syncthreads();
        for (int c = 0; c < cnt; ++c) {
            const float dx = xi - sx[c], dy = yi - sy[c], dz = zi - sz[c];
            const float d = dx * dx + dy * dy + dz * dz;
            if (d < bd[2]) {
                float dd = d; int ii = j0 + c;
#pragma unroll
                for (int s = 0; s < 3; ++s) {
                    if (dd < bd[s]) {
                        const float td = bd[s]; const int ti = bi[s];
                        bd[s] = dd; bi[s] = ii; dd = td; ii = ti;
                    }
                }
            }
        }
    }
    if (active) {
        float w0 = 1.0f / (bd[0] + 1e-8f);
        float w1 = 1.0f / (bd[1] + 1e-8f);
        float w2 = 1.0f / (bd[2] + 1e-8f);
        const float wsum = w0 + w1 + w2;
        const int p = b * n_dn + i;
        idx3[p * 3 + 0] = bi[0]; idx3[p * 3 + 1] = bi[1]; idx3[p * 3 + 2] = bi[2];
        w3[p * 3 + 0] = w0 / wsum; w3[p * 3 + 1] = w1 / wsum; w3[p * 3 + 2] = w2 / wsum;
    }
}

__global__ void k_interp3(const float* __restrict__ fsp, int n_sp, int C,
                          const int* __restrict__ idx3, const float* __restrict__ w3,
                          int n_dn, float* __restrict__ out)
{
    const int t = blockIdx.x * blockDim.x + threadIdx.x;
    if (t >= NB * n_dn * C) return;
    const int c = t % C;
    const int i = (t / C) % n_dn;
    const int b = t / (C * n_dn);
    const int p = b * n_dn + i;
    float acc = 0.0f;
#pragma unroll
    for (int s = 0; s < 3; ++s) {
        const int j = idx3[p * 3 + s];
        acc += w3[p * 3 + s] * fsp[((size_t)b * n_sp + j) * C + c];
    }
    out[t] = acc;
}

// dst[r, off+c] = src[r, c]
__global__ void k_copy_cols(float* __restrict__ dst, int dstStride, int off,
                            const float* __restrict__ src, int srcStride,
                            int rows, int C)
{
    const int t = blockIdx.x * blockDim.x + threadIdx.x;
    if (t >= rows * C) return;
    const int r = t / C, c = t % C;
    dst[(size_t)r * dstStride + off + c] = src[(size_t)r * srcStride + c];
}

// ---------------------------------------------------------------------------
// Host orchestration
// ---------------------------------------------------------------------------
static inline int ceil_div(int a, int b) { return (a + b - 1) / b; }

extern "C" void kernel_launch(void* const* d_in, const int* in_sizes, int n_in,
                              void* d_out, int out_size, void* d_ws, size_t ws_size,
                              hipStream_t stream)
{
    (void)in_sizes; (void)n_in; (void)out_size; (void)ws_size;

    const float* pts      = (const float*)d_in[0];
    const float* w_in     = (const float*)d_in[1];
    const float* w_bottle = (const float*)d_in[38];
    const float* w_dec[4] = {(const float*)d_in[39], (const float*)d_in[40],
                             (const float*)d_in[41], (const float*)d_in[42]};
    const float* w_head1  = (const float*)d_in[43];
    const float* w_head2  = (const float*)d_in[44];
    const float* b_head2  = (const float*)d_in[45];

    char* ws = (char*)d_ws;
    size_t off = 0;
    auto alloc = [&](size_t bytes) -> void* {
        void* p = ws + off;
        off += (bytes + 255) & ~(size_t)255;
        return p;
    };
    int*   IDX  = (int*)  alloc((size_t)NB * NPTS * 16 * sizeof(int));
    float* REL  = (float*)alloc((size_t)NB * NPTS * 16 * 10 * sizeof(float));
    float* XK   = (float*)alloc((size_t)NB * NPTS * 16 * 8  * sizeof(float));
    float* SC   = (float*)alloc((size_t)NB * NPTS * 16 * 8  * sizeof(float));
    float* AR0  = (float*)alloc((size_t)NB * NPTS * 64 * sizeof(float));
    float* AR1  = (float*)alloc((size_t)NB * NPTS * 64 * sizeof(float));
    float* AR2  = (float*)alloc((size_t)NB * NPTS * 64 * sizeof(float));
    float* F0   = (float*)alloc((size_t)NB * 8192 * 16  * sizeof(float));
    float* F1   = (float*)alloc((size_t)NB * 2048 * 32  * sizeof(float));
    float* F2   = (float*)alloc((size_t)NB * 512  * 64  * sizeof(float));
    float* F3   = (float*)alloc((size_t)NB * 128  * 128 * sizeof(float));
    int*   IDX3 = (int*)  alloc((size_t)NB * NPTS * 3 * sizeof(int));
    float* W3   = (float*)alloc((size_t)NB * NPTS * 3 * sizeof(float));

    auto linear = [&](const float* in, int inStride, const float* W, const float* bias,
                      float* out, int outStride, int outOff,
                      int R, int Cin, int Cout, int act, float scale) {
        k_linear_wmma<<<ceil_div(R, 16), 256, 0, stream>>>(
            in, inStride, W, bias, out, outStride, outOff, R, Cin, Cout, act, scale);
    };
    auto egrid = [&](int cnt) { return dim3(ceil_div(cnt, 256)); };

    const int ns[5]   = {8192, 2048, 512, 128, 32};
    const int dims[5] = {8, 16, 32, 64, 128};
    float* Fs[4] = {F0, F1, F2, F3};

    // x = lrelu(pts @ w_in.T * BN_S)  -> AR0 (B, 8192, 8)
    linear(pts, 6, w_in, nullptr, AR0, 8, 0, NB * NPTS, 6, 8, ACT_LRELU, BN_S);

    // ---------------- encoder ----------------
    for (int s = 0; s < 4; ++s) {
        const int n = ns[s], d = dims[s], dh = d / 2;
        const float* proj = (const float*)d_in[2 + s * 9 + 0];
        const float* pos1 = (const float*)d_in[2 + s * 9 + 1];
        const float* att1 = (const float*)d_in[2 + s * 9 + 2];
        const float* mlp1 = (const float*)d_in[2 + s * 9 + 3];
        const float* pos2 = (const float*)d_in[2 + s * 9 + 4];
        const float* att2 = (const float*)d_in[2 + s * 9 + 5];
        const float* mlp2 = (const float*)d_in[2 + s * 9 + 6];
        const float* wout = (const float*)d_in[2 + s * 9 + 7];
        const float* wsht = (const float*)d_in[2 + s * 9 + 8];
        const int Rp = NB * n;
        const int Rk = Rp * 16;

        k_knn16<<<dim3(ceil_div(n, 256), NB), 256, 0, stream>>>(pts, n, IDX);

        linear(AR0, d, proj, nullptr, AR1, dh, 0, Rp, d, dh, ACT_LRELU, 1.0f);

        k_build_rel<<<egrid(Rk), 256, 0, stream>>>(pts, IDX, n, REL);

        // LSE #1 + attention pool #1
        linear(REL, 10, pos1, nullptr, XK, d, 0, Rk, 10, dh, ACT_RELU, BN_S);
        k_concat_fe<<<egrid(Rk * dh), 256, 0, stream>>>(XK, d, dh, AR1, dh, Rk);
        linear(XK, d, att1, nullptr, SC, d, 0, Rk, d, d, ACT_NONE, 1.0f);
        k_softmax_pool<<<egrid(Rp * d), 256, 0, stream>>>(XK, SC, Rp, d, AR2);
        linear(AR2, d, mlp1, nullptr, AR1, dh, 0, Rp, d, dh, ACT_RELU, BN_S);

        // LSE #2 + attention pool #2
        linear(REL, 10, pos2, nullptr, XK, d, 0, Rk, 10, dh, ACT_RELU, BN_S);
        k_concat_fe<<<egrid(Rk * dh), 256, 0, stream>>>(XK, d, dh, AR1, dh, Rk);
        linear(XK, d, att2, nullptr, SC, d, 0, Rk, d, d, ACT_NONE, 1.0f);
        k_softmax_pool<<<egrid(Rp * d), 256, 0, stream>>>(XK, SC, Rp, d, AR2);
        linear(AR2, d, mlp2, nullptr, AR1, d, 0, Rp, d, d, ACT_RELU, BN_S);

        // main + shortcut -> lrelu -> Fs[s]
        linear(AR1, d, wout, nullptr, AR2, 2 * d, 0, Rp, d, 2 * d, ACT_NONE, BN_S);
        linear(AR0, d, wsht, nullptr, XK,  2 * d, 0, Rp, d, 2 * d, ACT_NONE, BN_S);
        k_add_lrelu<<<egrid(Rp * 2 * d), 256, 0, stream>>>(AR2, XK, Fs[s], Rp * 2 * d);

        k_subsample<<<egrid(NB * ns[s + 1] * 2 * d), 256, 0, stream>>>(
            Fs[s], AR0, n, ns[s + 1], 2 * d);
    }

    // ---------------- bottleneck (B, 32, 128) ----------------
    linear(AR0, 128, w_bottle, nullptr, AR1, 128, 0, NB * 32, 128, 128, ACT_RELU, BN_S);
    float* xbuf = AR1;

    // ---------------- decoder ----------------
    for (int i = 0; i < 4; ++i) {
        const int n_sp = ns[4 - i];
        const int n_dn = ns[3 - i];
        const int Cx   = dims[4 - i];
        const float* ef = Fs[3 - i];
        const int efC  = 2 * dims[3 - i];
        const int Cout = dims[3 - i];
        const int Rd   = NB * n_dn;

        k_find3<<<dim3(ceil_div(n_dn, 256), NB), 256, 0, stream>>>(pts, n_dn, n_sp, IDX3, W3);
        k_interp3<<<egrid(Rd * Cx), 256, 0, stream>>>(xbuf, n_sp, Cx, IDX3, W3, n_dn, AR2);
        k_copy_cols<<<egrid(Rd * Cx), 256, 0, stream>>>(AR0, Cx + efC, 0,  AR2, Cx,  Rd, Cx);
        k_copy_cols<<<egrid(Rd * efC), 256, 0, stream>>>(AR0, Cx + efC, Cx, ef,  efC, Rd, efC);
        linear(AR0, Cx + efC, w_dec[i], nullptr, AR1, Cout, 0, Rd, Cx + efC, Cout, ACT_RELU, BN_S);
        xbuf = AR1;
    }

    // ---------------- heads ----------------
    linear(xbuf, 8, w_head1, nullptr, AR0, 64, 0, NB * NPTS, 8, 64, ACT_RELU, BN_S);
    linear(AR0, 64, w_head2, b_head2, (float*)d_out, 13, 0, NB * NPTS, 64, 13, ACT_NONE, 1.0f);
}
